// MaskedAttn_20744692039971
// MI455X (gfx1250) — matrix operand order repaired and checked by reading, not compiled, not run on hardware
//
#include <hip/hip_runtime.h>
#include <hip/hip_bf16.h>
#include <hip/hip_fp16.h>

// ---------------------------------------------------------------------------
// MaskedAttn for MI455X (gfx1250, wave32, WMMA).
//   pre-passes: x->f16, qkv_w->f16, Wm[b]=proj_w*mask_proj[y[b]] -> f16
//   k1: qkv GEMM (all-f16 operands), V stored transposed
//   k2: attention (S=QK^T, softmax, O=PV) + fused mask_attn, f16 out
//   k3: projection vs. precomputed masked weights + bias, f32 out
// All GEMMs: v_wmma_f32_16x16x32_f16, 16x64 tile per wave (4 accumulators).
// All fragment loads are b128 vector loads; prefetch on streamed operands.
// ---------------------------------------------------------------------------

typedef __attribute__((ext_vector_type(16))) _Float16 v16h;
typedef __attribute__((ext_vector_type(8)))  _Float16 v8h;
typedef __attribute__((ext_vector_type(8)))  float    v8f;

#define BATCH    32
#define SEQ      197
#define CH       768
#define HEADS    12
#define HD       64
#define NPAD     224      // seq padded to multiple of 32 for the WMMA K dim
#define QT_TILES 13       // ceil(197/16) query row tiles
#define KT_TILES 14       // 224/16 key tiles (padded rows zeroed)
#define QKV_COLS 2304     // 3*C

__device__ __forceinline__ v8f wmma16(v16h a, v16h b, v8f c) {
  return __builtin_amdgcn_wmma_f32_16x16x32_f16(
      false, a, false, b, (short)0, c, false, false);
}

__device__ __forceinline__ v16h combine(v8h lo, v8h hh) {
  v16h r;
#pragma unroll
  for (int j = 0; j < 8; ++j) { r[j] = lo[j]; r[j + 8] = hh[j]; }
  return r;
}

// f16 A-fragment (16x32): lane row m, K = k0 + {hi*8+0..7, 16+hi*8+0..7}.
__device__ __forceinline__ v16h loadA_h(const _Float16* rowp, int k0, int hi) {
  v8h lo = *(const v8h*)(rowp + k0 + hi * 8);
  v8h hh = *(const v8h*)(rowp + k0 + 16 + hi * 8);
  return combine(lo, hh);
}

// f16 B-fragment (32x16): lane column n, K = k0 + {hi*8+0..7, 16+hi*8+0..7}.
__device__ __forceinline__ v16h loadB_h(const _Float16* colp, int k0, int hi) {
  v8h lo = *(const v8h*)(colp + k0 + hi * 8);
  v8h hh = *(const v8h*)(colp + k0 + 16 + hi * 8);
  return combine(lo, hh);
}

// ---------------------------------------------------------------------------
// Pre-pass A: f32 -> f16 elementwise (8 elements / thread, b128 in, b128 out)
// ---------------------------------------------------------------------------
__global__ void __launch_bounds__(256)
cvt_f32_to_f16_kernel(const float* __restrict__ src,
                      _Float16* __restrict__ dst, int n8) {
  const int t = blockIdx.x * 256 + threadIdx.x;
  if (t >= n8) return;
  const size_t o = (size_t)t * 8;
  float buf[8];
  *(float4*)(buf + 0) = *(const float4*)(src + o);
  *(float4*)(buf + 4) = *(const float4*)(src + o + 4);
  v8h r;
#pragma unroll
  for (int j = 0; j < 8; ++j) r[j] = (_Float16)buf[j];
  *(v8h*)(dst + o) = r;
}

// ---------------------------------------------------------------------------
// Pre-pass B: Wm16[b] = proj_w * mask_proj[y[b]] in f16 (8 elems / thread)
// ---------------------------------------------------------------------------
__global__ void __launch_bounds__(256)
masked_weight_kernel(const float* __restrict__ proj_w,
                     const float* __restrict__ mask_proj,
                     const int*   __restrict__ y,
                     _Float16*    __restrict__ wm16) {
  const int b = blockIdx.y;
  const int t = blockIdx.x * 256 + threadIdx.x;
  const size_t o = (size_t)t * 8;                    // within C*C
  const size_t mo = (size_t)y[b] * CH * CH + o;
  float w[8], mk[8];
  *(float4*)(w + 0)  = *(const float4*)(proj_w + o);
  *(float4*)(w + 4)  = *(const float4*)(proj_w + o + 4);
  *(float4*)(mk + 0) = *(const float4*)(mask_proj + mo);
  *(float4*)(mk + 4) = *(const float4*)(mask_proj + mo + 4);
  v8h r;
#pragma unroll
  for (int j = 0; j < 8; ++j) r[j] = (_Float16)(w[j] * mk[j]);
  *(v8h*)(wm16 + (size_t)b * CH * CH + o) = r;
}

// ---------------------------------------------------------------------------
// Kernel 1: qkv = x16 @ w16^T + qkv_b. One wave -> 16 rows x 64 cols
// (one (s,h) slice since col0 is 64-aligned). Q,K stored (B,H,NPAD,HD) f16;
// V stored transposed (B,H,HD,NPAD) f16. Padded rows (>=SEQ) written zero.
// ---------------------------------------------------------------------------
__global__ void __launch_bounds__(32)
qkv_gemm_kernel(const _Float16* __restrict__ x16,
                const _Float16* __restrict__ w16,
                const float* __restrict__ qkv_b,
                _Float16* __restrict__ qbuf,
                _Float16* __restrict__ kbuf,
                _Float16* __restrict__ vTbuf) {
  const int lane = threadIdx.x;
  const int m = lane & 15, hi = lane >> 4;
  const int b    = blockIdx.x / KT_TILES;
  const int nt   = blockIdx.x % KT_TILES;
  const int col0 = blockIdx.y * 64;

  const int row  = nt * 16 + m;
  const int arow = (row < SEQ) ? row : (SEQ - 1);
  const _Float16* xrow = x16 + (size_t)(b * SEQ + arow) * CH;
  const _Float16* wcol[4];
#pragma unroll
  for (int dt = 0; dt < 4; ++dt)
    wcol[dt] = w16 + (size_t)(col0 + dt * 16 + m) * CH;   // W^T col = W row

  v8f acc[4] = {};
  for (int k0 = 0; k0 < CH; k0 += 32) {
    __builtin_prefetch(xrow + k0 + 128, 0, 1);
    const v16h af = loadA_h(xrow, k0, hi);
#pragma unroll
    for (int dt = 0; dt < 4; ++dt) {
      __builtin_prefetch(wcol[dt] + k0 + 128, 0, 1);
      acc[dt] = wmma16(af, loadB_h(wcol[dt], k0, hi), acc[dt]);
    }
  }

  const int s = col0 / CH;
  const int h = (col0 % CH) / HD;
  const int bh = b * HEADS + h;
#pragma unroll
  for (int dt = 0; dt < 4; ++dt) {
    const int d = dt * 16 + m;
    const float bias = qkv_b[col0 + d];
#pragma unroll
    for (int r = 0; r < 8; ++r) {
      const int grow = nt * 16 + hi * 8 + r;
      const float val = (grow < SEQ) ? (acc[dt][r] + bias) : 0.f;
      if (s == 2)
        vTbuf[((size_t)bh * HD + d) * NPAD + grow] = (_Float16)val;
      else
        ((s == 0) ? qbuf : kbuf)[((size_t)bh * NPAD + grow) * HD + d] =
            (_Float16)val;
    }
  }
}

// ---------------------------------------------------------------------------
// Kernel 2: per (b,h,query-tile): S = (Q*scale)K^T -> softmax -> O = P V,
// fused per-class mask_attn scale, f16 output (B,NPAD,C).
// Softmax uses all 32 lanes (2 lanes per row) + shfl_xor cross-half combine.
// ---------------------------------------------------------------------------
__global__ void __launch_bounds__(32)
attn_kernel(const _Float16* __restrict__ qbuf,
            const _Float16* __restrict__ kbuf,
            const _Float16* __restrict__ vTbuf,
            const int*      __restrict__ y,
            const float*    __restrict__ mask_attn,
            _Float16*       __restrict__ obuf) {
  __shared__ float    sS[16][NPAD];   // f32 scores
  __shared__ _Float16 sP[16][NPAD];   // f16 probabilities

  const int lane = threadIdx.x;
  const int m = lane & 15, hi = lane >> 4;
  const int bh = blockIdx.x;
  const int b = bh / HEADS, h = bh % HEADS;
  const int qt = blockIdx.y;

  // Q fragments for the two K=32 chunks of D=64 (kept for the whole tile).
  const _Float16* qrow = qbuf + ((size_t)bh * NPAD + qt * 16 + m) * HD;
  const v16h qa0 = loadA_h(qrow, 0, hi);
  const v16h qa1 = loadA_h(qrow, 32, hi);

  // ---- scores: S[16 x NPAD] ----
  for (int jt = 0; jt < KT_TILES; ++jt) {
    const _Float16* kcol = kbuf + ((size_t)bh * NPAD + jt * 16 + m) * HD;
    __builtin_prefetch(kcol + 16 * HD, 0, 1);
    v8f s = {};
    s = wmma16(qa0, loadB_h(kcol, 0, hi), s);
    s = wmma16(qa1, loadB_h(kcol, 32, hi), s);
#pragma unroll
    for (int r = 0; r < 8; ++r)
      sS[hi * 8 + r][jt * 16 + m] = s[r] * 0.125f;   // SCALE = 64^-0.5
  }
  __syncthreads();

  // ---- softmax: row m, two lanes split columns [0,112) and [112,224) ----
  {
    const int c0 = hi * 112;
    const int c1 = (c0 + 112 < SEQ) ? (c0 + 112) : SEQ;
    float mx = -1e30f;
    for (int c = c0; c < c1; ++c) mx = fmaxf(mx, sS[m][c]);
    mx = fmaxf(mx, __shfl_xor(mx, 16, 32));
    float sum = 0.f;
    for (int c = c0; c < c1; ++c) {
      const float e = __expf(sS[m][c] - mx);
      sS[m][c] = e;
      sum += e;
    }
    sum += __shfl_xor(sum, 16, 32);
    const float inv = 1.f / sum;
    for (int c = c0; c < c1; ++c) sP[m][c] = (_Float16)(sS[m][c] * inv);
    for (int c = c1; c < c0 + 112; ++c) sP[m][c] = (_Float16)0.f;
  }
  __syncthreads();

  // ---- O = P @ V : 16x64 output, 4 accumulators; V^T is K-contiguous ----
  v8f o[4] = {};
  for (int kc = 0; kc < NPAD / 32; ++kc) {
    v16h pa = loadA_h(&sP[m][0], kc * 32, hi);      // ds_load_b128 x2
#pragma unroll
    for (int dt = 0; dt < 4; ++dt) {
      const _Float16* vcol = vTbuf + ((size_t)bh * HD + dt * 16 + m) * NPAD;
      o[dt] = wmma16(pa, loadB_h(vcol, kc * 32, hi), o[dt]);
    }
  }

  // fused per-class attention mask + f16 store in (B,NPAD,C) layout
  const int cls = y[b];
#pragma unroll
  for (int dt = 0; dt < 4; ++dt) {
    const int d = dt * 16 + m;
    const float scl = mask_attn[(size_t)cls * CH + h * HD + d];
#pragma unroll
    for (int r = 0; r < 8; ++r) {
      const int grow = qt * 16 + hi * 8 + r;
      obuf[((size_t)b * NPAD + grow) * CH + h * HD + d] =
          (_Float16)(o[dt][r] * scl);
    }
  }
}

// ---------------------------------------------------------------------------
// Kernel 3: out = o @ Wm16[b]^T + proj_b, Wm16 precomputed masked f16 weight.
// One wave -> 16 rows x 64 cols; pure f16 operand streams.
// ---------------------------------------------------------------------------
__global__ void __launch_bounds__(32)
proj_gemm_kernel(const _Float16* __restrict__ obuf,
                 const _Float16* __restrict__ wm16,
                 const float* __restrict__ proj_b,
                 float* __restrict__ out) {
  const int lane = threadIdx.x;
  const int m = lane & 15, hi = lane >> 4;
  const int b    = blockIdx.x / QT_TILES;
  const int nt   = blockIdx.x % QT_TILES;
  const int col0 = blockIdx.y * 64;

  const int row  = nt * 16 + m;
  const int arow = (row < SEQ) ? row : (SEQ - 1);
  const _Float16* oa = obuf + ((size_t)b * NPAD + arow) * CH;
  const _Float16* wc[4];
#pragma unroll
  for (int dt = 0; dt < 4; ++dt)
    wc[dt] = wm16 + ((size_t)b * CH + col0 + dt * 16 + m) * CH;

  v8f acc[4] = {};
  for (int k0 = 0; k0 < CH; k0 += 32) {
    __builtin_prefetch(oa + k0 + 128, 0, 1);
    const v16h af = loadA_h(oa, k0, hi);
#pragma unroll
    for (int dt = 0; dt < 4; ++dt) {
      __builtin_prefetch(wc[dt] + k0 + 128, 0, 1);
      acc[dt] = wmma16(af, loadB_h(wc[dt], k0, hi), acc[dt]);
    }
  }

#pragma unroll
  for (int dt = 0; dt < 4; ++dt) {
    const int col = col0 + dt * 16 + m;
    const float bias = proj_b[col];
#pragma unroll
    for (int r = 0; r < 8; ++r) {
      const int grow = nt * 16 + hi * 8 + r;
      if (grow < SEQ)
        out[((size_t)b * SEQ + grow) * CH + col] = acc[dt][r] + bias;
    }
  }
}

// ---------------------------------------------------------------------------
extern "C" void kernel_launch(void* const* d_in, const int* in_sizes, int n_in,
                              void* d_out, int out_size, void* d_ws, size_t ws_size,
                              hipStream_t stream) {
  const float* x         = (const float*)d_in[0];
  const int*   y         = (const int*)  d_in[1];
  const float* qkv_w     = (const float*)d_in[2];
  const float* qkv_b     = (const float*)d_in[3];
  const float* proj_w    = (const float*)d_in[4];
  const float* proj_b    = (const float*)d_in[5];
  const float* mask_attn = (const float*)d_in[6];
  const float* mask_proj = (const float*)d_in[7];
  float* out = (float*)d_out;

  // Workspace layout (halfs):
  //   q/k (B,H,NPAD,HD), vT (B,H,HD,NPAD), o (B,NPAD,C),
  //   x16 (B,SEQ,C), w16 (3C,C), wm16 (B,C,C)     total ~95 MB
  const size_t perQKV = (size_t)BATCH * HEADS * NPAD * HD;
  const size_t nX  = (size_t)BATCH * SEQ * CH;        // 4,841,472
  const size_t nW  = (size_t)QKV_COLS * CH;           // 1,769,472
  const size_t nWm = (size_t)BATCH * CH * CH;         // 18,874,368
  _Float16* qbuf  = (_Float16*)d_ws;
  _Float16* kbuf  = qbuf + perQKV;
  _Float16* vTbuf = kbuf + perQKV;
  _Float16* obuf  = vTbuf + perQKV;
  _Float16* x16   = obuf + perQKV * HEADS / HEADS + ((size_t)BATCH * NPAD * CH - perQKV);  // after obuf
  // simpler: recompute offsets explicitly
  x16 = ((_Float16*)d_ws) + 3 * perQKV + (size_t)BATCH * NPAD * CH;
  _Float16* w16   = x16 + nX;
  _Float16* wm16  = w16 + nW;

  // pre-passes (x and qkv_w sizes are multiples of 8)
  cvt_f32_to_f16_kernel<<<dim3((nX / 8 + 255) / 256), 256, 0, stream>>>(
      x, x16, (int)(nX / 8));
  cvt_f32_to_f16_kernel<<<dim3((nW / 8 + 255) / 256), 256, 0, stream>>>(
      qkv_w, w16, (int)(nW / 8));
  masked_weight_kernel<<<dim3((CH * CH) / (256 * 8), BATCH), 256, 0, stream>>>(
      proj_w, mask_proj, y, wm16);

  qkv_gemm_kernel<<<dim3(BATCH * KT_TILES, QKV_COLS / 64), 32, 0, stream>>>(
      x16, w16, qkv_b, qbuf, kbuf, vTbuf);
  attn_kernel<<<dim3(BATCH * HEADS, QT_TILES), 32, 0, stream>>>(
      qbuf, kbuf, vTbuf, y, mask_attn, obuf);
  proj_gemm_kernel<<<dim3(BATCH * QT_TILES, CH / 64), 32, 0, stream>>>(
      obuf, wm16, proj_b, out);
}
